// HeteroGINEEvolveGCN_82952998355883
// MI455X (gfx1250) — compile-verified, matrix operand order checked by reading, not run on hardware
//
#include <hip/hip_runtime.h>
#include <math.h>

typedef __attribute__((ext_vector_type(16))) _Float16 v16h;
typedef __attribute__((ext_vector_type(8)))  _Float16 v8h;
typedef __attribute__((ext_vector_type(8)))  float    v8f;

namespace {
constexpr int Tn = 6, Nn = 20000, En = 160000, Bn = 128;
constexpr int DNn = 64, DEn = 32, Hn = 128, Cn = 2, Ln = 3;
constexpr float BN_EPS = 1e-5f;
}

__device__ __forceinline__ float sigmoidf_(float x) { return 1.f / (1.f + __expf(-x)); }

// ---------------------------------------------------------------------------
// WMMA GEMM: Y[M,NO] = act(X[M,K] @ W + bias)
// Block = 128 threads (4 waves) -> 64(M) x 64(N) output tile.
// The 64-column weight panel is staged into LDS *pre-converted to f16*,
// transposed to [n][k] with +8-half pad (distinct banks for the 16 fragment
// lanes, 16B-aligned ds_load_b128 reads).
//   TRANS==0: W is (K,NO) row-major.   TRANS==1: W is (NO,K) row-major (W^T).
// Each wave owns one 16-row strip and 4 N-tiles: one A fragment
// (4x global_load_b128 + 8x v_cvt_pk_f16_f32) feeds 4 WMMAs per 32-K chunk;
// B fragments come straight from LDS as f16 (2x ds_load_b128 each).
// M multiple of 16 (tail waves masked), K multiple of 32, NO multiple of 64.
// ---------------------------------------------------------------------------
template <int TRANS, int RELU>
__global__ void gemm_wmma_kernel(const float* __restrict__ X, const float* __restrict__ W,
                                 const float* __restrict__ bias, float* __restrict__ Y,
                                 int M, int K, int NO)
{
  extern __shared__ _Float16 lw[];         // 64 x (K+8) halfs
  const int wave = threadIdx.x >> 5;
  const int lane = threadIdx.x & 31;
  const int m0 = blockIdx.x * 64 + wave * 16;
  const int n0 = blockIdx.y * 64;
  const int half = lane >> 4;              // lanes 0-15 vs 16-31 (CDNA5 K split)
  const int l16  = lane & 15;
  const int KP = K + 8;                    // pad: bank spread + 16B alignment

  // ---- cooperative stage of the 64-wide weight panel, f32 -> f16 ----
  if (TRANS) {
    for (int n = 0; n < 64; ++n)
      for (int k = threadIdx.x; k < K; k += 128)
        lw[n * KP + k] = (_Float16)W[(long)(n0 + n) * K + k];
  } else {
    for (int idx = threadIdx.x; idx < K * 64; idx += 128) {
      int k = idx >> 6, n = idx & 63;
      lw[n * KP + k] = (_Float16)W[(long)k * NO + (n0 + n)];
    }
  }
  __syncthreads();

  if (m0 < M) {                            // wave-uniform; EXEC all-1s inside
    const long rowA = (long)(m0 + l16) * K;
    v8f acc[4] = {v8f{}, v8f{}, v8f{}, v8f{}};
    for (int k0 = 0; k0 < K; k0 += 32) {
      const int kaL = k0 + half * 8;       // low  8-run: K in {0..7} / {8..15}
      const int kaH = kaL + 16;            // high 8-run: K in {16..23} / {24..31}
      const float4 a0 = *(const float4*)&X[rowA + kaL];
      const float4 a1 = *(const float4*)&X[rowA + kaL + 4];
      const float4 a2 = *(const float4*)&X[rowA + kaH];
      const float4 a3 = *(const float4*)&X[rowA + kaH + 4];
      if (k0 + 32 < K) __builtin_prefetch(&X[rowA + kaL + 32], 0, 0);
      v16h a;
      a[0]=(_Float16)a0.x; a[1]=(_Float16)a0.y; a[2]=(_Float16)a0.z; a[3]=(_Float16)a0.w;
      a[4]=(_Float16)a1.x; a[5]=(_Float16)a1.y; a[6]=(_Float16)a1.z; a[7]=(_Float16)a1.w;
      a[8]=(_Float16)a2.x; a[9]=(_Float16)a2.y; a[10]=(_Float16)a2.z; a[11]=(_Float16)a2.w;
      a[12]=(_Float16)a3.x; a[13]=(_Float16)a3.y; a[14]=(_Float16)a3.z; a[15]=(_Float16)a3.w;
#pragma unroll
      for (int nt = 0; nt < 4; ++nt) {
        const _Float16* bp = &lw[(nt * 16 + l16) * KP];
        const v8h blo = *(const v8h*)&bp[kaL];   // ds_load_b128
        const v8h bhi = *(const v8h*)&bp[kaH];   // ds_load_b128
        const v16h b = __builtin_shufflevector(blo, bhi,
            0,1,2,3,4,5,6,7,8,9,10,11,12,13,14,15);
        acc[nt] = __builtin_amdgcn_wmma_f32_16x16x32_f16(false, a, false, b,
                                                         (short)0, acc[nt], false, false);
      }
    }
#pragma unroll
    for (int nt = 0; nt < 4; ++nt) {
      const int n = n0 + nt * 16 + l16;
      const float bnv = bias[n];
#pragma unroll
      for (int r = 0; r < 8; ++r) {
        // C layout: VGPR r -> M = r (lanes 0-15) / 8+r (lanes 16-31); N = lane%16
        const int m = m0 + half * 8 + r;
        float v = acc[nt][r] + bnv;
        if (RELU) v = fmaxf(v, 0.f);
        Y[(long)m * NO + n] = v;
      }
    }
  }
}

// ---------------------------------------------------------------------------
__global__ void fill_zero_kernel(float* __restrict__ p, long n)
{
  long i = (long)blockIdx.x * blockDim.x + threadIdx.x;
  if (i < n) p[i] = 0.f;
}

// Per-column batch statistics (training-mode BatchNorm1d, biased variance).
__global__ void col_stats_kernel(const float* __restrict__ X, float* __restrict__ mean,
                                 float* __restrict__ var, int M, int NO)
{
  const int j = blockIdx.x;
  __shared__ float ssum[256];
  __shared__ float ssq[256];
  float s = 0.f, q = 0.f;
  for (int i = threadIdx.x; i < M; i += 256) {
    float v = X[(long)i * NO + j];
    s += v; q += v * v;
  }
  ssum[threadIdx.x] = s; ssq[threadIdx.x] = q;
  __syncthreads();
  for (int st = 128; st > 0; st >>= 1) {
    if (threadIdx.x < st) { ssum[threadIdx.x] += ssum[threadIdx.x + st]; ssq[threadIdx.x] += ssq[threadIdx.x + st]; }
    __syncthreads();
  }
  if (threadIdx.x == 0) {
    float m = ssum[0] / (float)M;
    mean[j] = m;
    var[j]  = ssq[0] / (float)M - m * m;
  }
}

__global__ void bn_relu_kernel(const float* __restrict__ X, const float* __restrict__ mean,
                               const float* __restrict__ var, const float* __restrict__ g,
                               const float* __restrict__ beta, float* __restrict__ Y,
                               long n, int NO)
{
  long i = (long)blockIdx.x * blockDim.x + threadIdx.x;
  if (i >= n) return;
  int j = (int)(i % NO);
  float v = g[j] * (X[i] - mean[j]) * rsqrtf(var[j] + BN_EPS) + beta[j];
  Y[i] = fmaxf(v, 0.f);
}

// h = a*relu(bn(z)) + (1-a)*h   with a = sigmoid(res_w[li])
__global__ void bn_relu_residual_kernel(const float* __restrict__ Z, const float* __restrict__ mean,
                                        const float* __restrict__ var, const float* __restrict__ g,
                                        const float* __restrict__ beta, const float* __restrict__ resw,
                                        int li, float* __restrict__ Hbuf, long n, int NO)
{
  long i = (long)blockIdx.x * blockDim.x + threadIdx.x;
  if (i >= n) return;
  int j = (int)(i % NO);
  float a = sigmoidf_(resw[li]);
  float hn = fmaxf(g[j] * (Z[i] - mean[j]) * rsqrtf(var[j] + BN_EPS) + beta[j], 0.f);
  Hbuf[i] = a * hn + (1.f - a) * Hbuf[i];
}

// z = (1+eps)*h + agg
__global__ void eps_combine_kernel(const float* __restrict__ Hbuf, const float* __restrict__ agg,
                                   const float* __restrict__ epsp, float* __restrict__ Z, long n)
{
  long i = (long)blockIdx.x * blockDim.x + threadIdx.x;
  if (i >= n) return;
  Z[i] = (1.f + epsp[0]) * Hbuf[i] + agg[i];
}

// Per-edge message m = relu(h[src] + e), scatter-sum into agg[dst] (f32 atomics).
__global__ void edge_msg_scatter_kernel(const float* __restrict__ Hbuf, const float* __restrict__ Ee,
                                        const int* __restrict__ src, const int* __restrict__ dst,
                                        float* __restrict__ agg)
{
  long idx = (long)blockIdx.x * blockDim.x + threadIdx.x;
  const long total = (long)En * (Hn / 4);
  if (idx >= total) return;
  int e  = (int)(idx / (Hn / 4));
  int j4 = (int)(idx % (Hn / 4)) * 4;
  int s = src[e], d = dst[e];
  const float4 hv = *(const float4*)&Hbuf[(long)s * Hn + j4];
  const float4 ev = *(const float4*)&Ee[(long)e * Hn + j4];
  float* ag = &agg[(long)d * Hn + j4];
  atomicAdd(ag + 0, fmaxf(hv.x + ev.x, 0.f));
  atomicAdd(ag + 1, fmaxf(hv.y + ev.y, 0.f));
  atomicAdd(ag + 2, fmaxf(hv.z + ev.z, 0.f));
  atomicAdd(ag + 3, fmaxf(hv.w + ev.w, 0.f));
}

// gate[i] = G1[i,:] . W2 + b2   (G1 is N x H/2, already ReLU'd)
__global__ void gate_reduce_kernel(const float* __restrict__ G1, const float* __restrict__ W2,
                                   const float* __restrict__ b2, float* __restrict__ gate, int M)
{
  int i = blockIdx.x * blockDim.x + threadIdx.x;
  if (i >= M) return;
  float s = b2[0];
  const float* row = &G1[(long)i * (Hn / 2)];
#pragma unroll 8
  for (int j = 0; j < Hn / 2; ++j) s += row[j] * W2[j];
  gate[i] = s;
}

// batch is sorted: record [start,end) per graph into segs[0..B) / segs[B..2B)
__global__ void seg_ranges_kernel(const int* __restrict__ batch, int* __restrict__ segs)
{
  int i = blockIdx.x * blockDim.x + threadIdx.x;
  if (i >= Nn) return;
  int b = batch[i];
  if (i == 0) segs[b] = 0;
  else { int pb = batch[i - 1]; if (pb != b) { segs[b] = i; segs[Bn + pb] = i; } }
  if (i == Nn - 1) segs[Bn + b] = Nn;
}

__global__ void seg_softmax_stats_kernel(const float* __restrict__ gate, const int* __restrict__ segs,
                                         float* __restrict__ gmax, float* __restrict__ gsum)
{
  const int b = blockIdx.x;
  const int s = segs[b], e = segs[Bn + b];
  __shared__ float red[256];
  float mx = -3.4e38f;
  for (int i = s + threadIdx.x; i < e; i += 256) mx = fmaxf(mx, gate[i]);
  red[threadIdx.x] = mx; __syncthreads();
  for (int st = 128; st > 0; st >>= 1) {
    if (threadIdx.x < st) red[threadIdx.x] = fmaxf(red[threadIdx.x], red[threadIdx.x + st]);
    __syncthreads();
  }
  const float m = red[0]; __syncthreads();
  float sum = 0.f;
  for (int i = s + threadIdx.x; i < e; i += 256) sum += __expf(gate[i] - m);
  red[threadIdx.x] = sum; __syncthreads();
  for (int st = 128; st > 0; st >>= 1) {
    if (threadIdx.x < st) red[threadIdx.x] += red[threadIdx.x + st];
    __syncthreads();
  }
  if (threadIdx.x == 0) { gmax[b] = m; gsum[b] = (red[0] > 0.f) ? red[0] : 1.f; }
}

// embs[t,b,:] = sum_i softmax_w(i) * feat[i,:]
__global__ void att_pool_kernel(const float* __restrict__ gate, const float* __restrict__ feat,
                                const int* __restrict__ segs, const float* __restrict__ gmax,
                                const float* __restrict__ gsum, float* __restrict__ embs, int t)
{
  const int b = blockIdx.x;
  const int j = threadIdx.x;               // H threads
  const int s = segs[b], e = segs[Bn + b];
  const float m = gmax[b], inv = 1.f / gsum[b];
  float acc = 0.f;
  for (int i = s; i < e; ++i) {
    float w = __expf(gate[i] - m) * inv;
    acc += w * feat[(long)i * Hn + j];
  }
  embs[((long)t * Bn + b) * Hn + j] = acc;
}

// LSTM recurrence: block per batch row, thread j owns hidden unit j (4 gates).
// Xg = x @ Wih^T + b, precomputed for all T (shape T,B,4H). h lives in LDS.
__global__ void lstm_seq_kernel(const float* __restrict__ Xg, const float* __restrict__ Whh,
                                float* __restrict__ Yout, float* __restrict__ Hfinal,
                                int reverse, int outOffset, int writeY, int writeH)
{
  const int b = blockIdx.x;
  const int j = threadIdx.x;               // H threads
  __shared__ float hs[Hn];
  float c = 0.f;
  hs[j] = 0.f;
  __syncthreads();
  for (int t = 0; t < Tn; ++t) {
    const int tt = reverse ? (Tn - 1 - t) : t;
    const float* xg = &Xg[((long)tt * Bn + b) * (4 * Hn)];
    float zi = xg[j], zf = xg[Hn + j], zg = xg[2 * Hn + j], zo = xg[3 * Hn + j];
    const float* wi = &Whh[(long)(0 * Hn + j) * Hn];
    const float* wf = &Whh[(long)(1 * Hn + j) * Hn];
    const float* wg = &Whh[(long)(2 * Hn + j) * Hn];
    const float* wo = &Whh[(long)(3 * Hn + j) * Hn];
#pragma unroll 4
    for (int k = 0; k < Hn; ++k) {
      float hk = hs[k];
      zi += wi[k] * hk; zf += wf[k] * hk; zg += wg[k] * hk; zo += wo[k] * hk;
    }
    c = sigmoidf_(zf) * c + sigmoidf_(zi) * tanhf(zg);
    float hnew = sigmoidf_(zo) * tanhf(c);
    __syncthreads();
    hs[j] = hnew;
    __syncthreads();
    if (writeY) Yout[((long)tt * Bn + b) * (2 * Hn) + outOffset + j] = hnew;
  }
  if (writeH) Hfinal[(long)b * (2 * Hn) + outOffset + j] = hs[j];
}

// out[b,c] = relu'd temporal features @ cls_W + cls_b
__global__ void cls_kernel(const float* __restrict__ Tf, const float* __restrict__ clsW,
                           const float* __restrict__ clsb, float* __restrict__ out)
{
  const int b = blockIdx.x;
  const int ci = threadIdx.x;              // C threads
  float s = clsb[ci];
  for (int k = 0; k < Hn; ++k) s += Tf[(long)b * Hn + k] * clsW[k * Cn + ci];
  out[b * Cn + ci] = s;
}

// ---------------------------------------------------------------------------
template <int TRANS, int RELU>
static inline void launch_gemm_t(const float* X, const float* W, const float* b, float* Y,
                                 int M, int K, int NO, hipStream_t s)
{
  dim3 grid((M + 63) / 64, NO / 64);
  size_t shmem = (size_t)64 * (K + 8) * sizeof(_Float16);
  gemm_wmma_kernel<TRANS, RELU><<<grid, dim3(128), shmem, s>>>(X, W, b, Y, M, K, NO);
}
static inline void launch_fill(float* p, long n, hipStream_t s)
{
  fill_zero_kernel<<<(int)((n + 255) / 256), 256, 0, s>>>(p, n);
}

extern "C" void kernel_launch(void* const* d_in, const int* in_sizes, int n_in,
                              void* d_out, int out_size, void* d_ws, size_t ws_size,
                              hipStream_t stream)
{
  (void)in_sizes; (void)n_in; (void)out_size; (void)ws_size;
  // --- inputs (JAX pytree sorted-key flatten of setup_inputs() dict) ---
  const int*   batch  = (const int*)  d_in[0];
  const float* ea_seq = (const float*)d_in[1];   // (T,E,DE)
  const int*   ei_seq = (const int*)  d_in[2];   // (T,2,E)
  const float* att_W  = (const float*)d_in[3];
  const float* att_b  = (const float*)d_in[4];
  const float* cls_W  = (const float*)d_in[5];
  const float* cls_b  = (const float*)d_in[6];
  // convs: base 7 + 13*li; order: W1,W2,We,b1,b2,be,beta1,beta2,beta3,eps,g1,g2,g3
  const float* gate_W1 = (const float*)d_in[46];
  const float* gate_W2 = (const float*)d_in[47];
  const float* gate_b1 = (const float*)d_in[48];
  const float* gate_b2 = (const float*)d_in[49];
  const float* in_W    = (const float*)d_in[50];
  const float* in_b    = (const float*)d_in[51];
  const float* in_beta = (const float*)d_in[52];
  const float* in_g    = (const float*)d_in[53];
  // lstm cells: base 54 + 3*c; order: Whh, Wih, b; cells [l0f,l0b,l1f,l1b]
  const float* out_W  = (const float*)d_in[66];
  const float* out_b  = (const float*)d_in[67];
  const float* res_w  = (const float*)d_in[68];
  const float* x_seq  = (const float*)d_in[69];  // (T,N,DN)
  float* out = (float*)d_out;                    // (B,C)

  // --- workspace carve ---
  float* w = (float*)d_ws;
  float* hbuf = w;  w += (size_t)Nn * Hn;
  float* zbuf = w;  w += (size_t)Nn * Hn;
  float* z2buf= w;  w += (size_t)Nn * Hn;
  float* agg  = w;  w += (size_t)Nn * Hn;
  float* eenc = w;  w += (size_t)En * Hn;
  float* g1b  = w;  w += (size_t)Nn * (Hn / 2);
  float* gate = w;  w += (size_t)Nn;
  float* feat = w;  w += (size_t)Nn * Hn;
  float* embs = w;  w += (size_t)Tn * Bn * Hn;
  float* ybuf = w;  w += (size_t)Tn * Bn * 2 * Hn;
  float* xg   = w;  w += (size_t)Tn * Bn * 4 * Hn;
  float* temporal = w; w += (size_t)Bn * 2 * Hn;
  float* tfeat    = w; w += (size_t)Bn * Hn;
  float* meanb = w; w += Hn;
  float* varb  = w; w += Hn;
  float* gmax  = w; w += Bn;
  float* gsum  = w; w += Bn;
  int*   segs  = (int*)w;            // 2*Bn ints

  const long NH = (long)Nn * Hn;

  // segment ranges (batch constant across snapshots)
  launch_fill((float*)segs, 2 * Bn, stream);
  seg_ranges_kernel<<<(Nn + 255) / 256, 256, 0, stream>>>(batch, segs);

  for (int t = 0; t < Tn; ++t) {
    const float* xt  = x_seq + (size_t)t * Nn * DNn;
    const float* eat = ea_seq + (size_t)t * En * DEn;
    const int* src = ei_seq + (size_t)t * 2 * En;
    const int* dst = src + En;

    // input projection + BN + ReLU
    launch_gemm_t<0, 0>(xt, in_W, in_b, zbuf, Nn, DNn, Hn, stream);
    col_stats_kernel<<<Hn, 256, 0, stream>>>(zbuf, meanb, varb, Nn, Hn);
    bn_relu_kernel<<<(int)((NH + 255) / 256), 256, 0, stream>>>(zbuf, meanb, varb, in_g, in_beta, hbuf, NH, Hn);

    for (int li = 0; li < Ln; ++li) {
      const int cb = 7 + 13 * li;
      const float* W1 = (const float*)d_in[cb + 0];
      const float* W2 = (const float*)d_in[cb + 1];
      const float* We = (const float*)d_in[cb + 2];
      const float* b1 = (const float*)d_in[cb + 3];
      const float* b2 = (const float*)d_in[cb + 4];
      const float* be = (const float*)d_in[cb + 5];
      const float* beta1 = (const float*)d_in[cb + 6];
      const float* beta2 = (const float*)d_in[cb + 7];
      const float* beta3 = (const float*)d_in[cb + 8];
      const float* epsp  = (const float*)d_in[cb + 9];
      const float* g1 = (const float*)d_in[cb + 10];
      const float* g2 = (const float*)d_in[cb + 11];
      const float* g3 = (const float*)d_in[cb + 12];

      // edge encoder (WMMA GEMM E x 32 -> 128)
      launch_gemm_t<0, 0>(eat, We, be, eenc, En, DEn, Hn, stream);
      // scatter-sum of relu(h[src]+e) into agg[dst]
      launch_fill(agg, NH, stream);
      {
        long tot = (long)En * (Hn / 4);
        edge_msg_scatter_kernel<<<(int)((tot + 255) / 256), 256, 0, stream>>>(hbuf, eenc, src, dst, agg);
      }
      eps_combine_kernel<<<(int)((NH + 255) / 256), 256, 0, stream>>>(hbuf, agg, epsp, zbuf, NH);
      // MLP: two GEMM+BN+ReLU
      launch_gemm_t<0, 0>(zbuf, W1, b1, z2buf, Nn, Hn, Hn, stream);
      col_stats_kernel<<<Hn, 256, 0, stream>>>(z2buf, meanb, varb, Nn, Hn);
      bn_relu_kernel<<<(int)((NH + 255) / 256), 256, 0, stream>>>(z2buf, meanb, varb, g1, beta1, zbuf, NH, Hn);
      launch_gemm_t<0, 0>(zbuf, W2, b2, z2buf, Nn, Hn, Hn, stream);
      col_stats_kernel<<<Hn, 256, 0, stream>>>(z2buf, meanb, varb, Nn, Hn);
      bn_relu_kernel<<<(int)((NH + 255) / 256), 256, 0, stream>>>(z2buf, meanb, varb, g2, beta2, zbuf, NH, Hn);
      // block BN + ReLU + gated residual
      col_stats_kernel<<<Hn, 256, 0, stream>>>(zbuf, meanb, varb, Nn, Hn);
      bn_relu_residual_kernel<<<(int)((NH + 255) / 256), 256, 0, stream>>>(zbuf, meanb, varb, g3, beta3, res_w, li, hbuf, NH, Hn);
    }

    // GlobalAttention readout
    launch_gemm_t<0, 1>(hbuf, gate_W1, gate_b1, g1b, Nn, Hn, Hn / 2, stream);
    gate_reduce_kernel<<<(Nn + 255) / 256, 256, 0, stream>>>(g1b, gate_W2, gate_b2, gate, Nn);
    launch_gemm_t<0, 1>(hbuf, att_W, att_b, feat, Nn, Hn, Hn, stream);
    seg_softmax_stats_kernel<<<Bn, 256, 0, stream>>>(gate, segs, gmax, gsum);
    att_pool_kernel<<<Bn, Hn, 0, stream>>>(gate, feat, segs, gmax, gsum, embs, t);
  }

  // --- 2-layer bidirectional LSTM over snapshot embeddings ---
  // layer 0 forward (cell 0): y[:, :, 0:H]
  launch_gemm_t<1, 0>(embs, (const float*)d_in[54 + 1], (const float*)d_in[54 + 2], xg, Tn * Bn, Hn, 4 * Hn, stream);
  lstm_seq_kernel<<<Bn, Hn, 0, stream>>>(xg, (const float*)d_in[54 + 0], ybuf, nullptr, 0, 0, 1, 0);
  // layer 0 backward (cell 1): y[:, :, H:2H] (written at original time index == yb[::-1])
  launch_gemm_t<1, 0>(embs, (const float*)d_in[57 + 1], (const float*)d_in[57 + 2], xg, Tn * Bn, Hn, 4 * Hn, stream);
  lstm_seq_kernel<<<Bn, Hn, 0, stream>>>(xg, (const float*)d_in[57 + 0], ybuf, nullptr, 1, Hn, 1, 0);
  // layer 1 forward (cell 2): final hidden -> temporal[:, 0:H]
  launch_gemm_t<1, 0>(ybuf, (const float*)d_in[60 + 1], (const float*)d_in[60 + 2], xg, Tn * Bn, 2 * Hn, 4 * Hn, stream);
  lstm_seq_kernel<<<Bn, Hn, 0, stream>>>(xg, (const float*)d_in[60 + 0], nullptr, temporal, 0, 0, 0, 1);
  // layer 1 backward (cell 3): final hidden -> temporal[:, H:2H]
  launch_gemm_t<1, 0>(ybuf, (const float*)d_in[63 + 1], (const float*)d_in[63 + 2], xg, Tn * Bn, 2 * Hn, 4 * Hn, stream);
  lstm_seq_kernel<<<Bn, Hn, 0, stream>>>(xg, (const float*)d_in[63 + 0], nullptr, temporal, 1, Hn, 0, 1);

  // output head
  launch_gemm_t<0, 1>(temporal, out_W, out_b, tfeat, Bn, 2 * Hn, Hn, stream);
  cls_kernel<<<Bn, Cn, 0, stream>>>(tfeat, cls_W, cls_b, out);
}
